// TrajectoryDecoder_8211977470336
// MI455X (gfx1250) — compile-verified, matrix operand order checked by reading, not run on hardware
//
#include <hip/hip_runtime.h>
#include <hip/hip_bf16.h>

#define HORIZON  60
#define HID      128
#define DIN      256
#define G3       384      // 3*HID
#define BTILE    64       // batch rows per block
#define NTHREADS 128      // 4 waves, one 16-row strip each

typedef _Float16 v8h  __attribute__((ext_vector_type(8)));
typedef _Float16 v16h __attribute__((ext_vector_type(16)));
typedef float    v8f  __attribute__((ext_vector_type(8)));

__device__ __forceinline__ v8f wmma_f32_f16(v16h a, v16h b, v8f c) {
  // D = A(16x32 f16) x B(32x16 f16) + C(16x16 f32)
  return __builtin_amdgcn_wmma_f32_16x16x32_f16(false, a, false, b, (short)0, c,
                                                false, false);
}

// A/B fragment (16-bit, K=32 slab) from an LDS row (contiguous K, f16).
// lane<16: halves {k0+koff..+7} and {k0+16+koff..+7} with koff=0; lane>=16: koff=8.
__device__ __forceinline__ v16h frag_lds(const _Float16* rowp, int k0, int koff) {
  v8h lo = *(const v8h*)(rowp + k0 + koff);
  v8h hi = *(const v8h*)(rowp + k0 + 16 + koff);
  v16h r;
#pragma unroll
  for (int i = 0; i < 8; ++i) { r[i] = lo[i]; r[i + 8] = hi[i]; }
  return r;
}

// Same fragment built from a global f32 row (convert on the fly).
__device__ __forceinline__ v16h frag_gf32(const float* rowp, int k0, int koff) {
  const float4* p0 = (const float4*)(rowp + k0 + koff);
  const float4* p1 = (const float4*)(rowp + k0 + 16 + koff);
  float4 a0 = p0[0], a1 = p0[1], b0 = p1[0], b1 = p1[1];
  v16h r;
  r[0]  = (_Float16)a0.x; r[1]  = (_Float16)a0.y; r[2]  = (_Float16)a0.z; r[3]  = (_Float16)a0.w;
  r[4]  = (_Float16)a1.x; r[5]  = (_Float16)a1.y; r[6]  = (_Float16)a1.z; r[7]  = (_Float16)a1.w;
  r[8]  = (_Float16)b0.x; r[9]  = (_Float16)b0.y; r[10] = (_Float16)b0.z; r[11] = (_Float16)b0.w;
  r[12] = (_Float16)b1.x; r[13] = (_Float16)b1.y; r[14] = (_Float16)b1.z; r[15] = (_Float16)b1.w;
  return r;
}

__device__ __forceinline__ float sigmoid_(float x) {
  x = fminf(fmaxf(x, -30.f), 30.f);
  return 1.f / (1.f + __expf(-x));
}
__device__ __forceinline__ float tanh_(float x) {
  x = fminf(fmaxf(x, -15.f), 15.f);
  float e = __expf(-2.f * x);
  return (1.f - e) / (1.f + e);
}

__global__ void __launch_bounds__(NTHREADS, 1)
gru_traj_kernel(const float* __restrict__ x,     const float* __restrict__ W_ih,
                const float* __restrict__ W_hh,  const float* __restrict__ b_ih,
                const float* __restrict__ b_hh,  const float* __restrict__ W_out,
                const float* __restrict__ b_out, float* __restrict__ out) {
  extern __shared__ _Float16 smem[];
  _Float16* sWhh = smem;                    // [384][128] f16 row-major (row = gate col)
  _Float16* sGpT = smem + G3 * HID;         // [384][64]  f16 TRANSPOSED gp: [col][row]
  _Float16* sH   = sGpT + G3 * BTILE;       // [64][128]  f16 hidden state, row-major

  const int tid   = threadIdx.x;
  const int lane  = tid & 31;
  const int w     = tid >> 5;               // wave id 0..3 -> 16-row strip
  const int nl    = lane & 15;              // row (A) / col (B,C,D) index
  const int koff  = (lane >> 4) << 3;       // 0 or 8 : K sub-offset AND C/D row +8
  const int lr0   = w * 16;                 // local strip base row
  const int bRow0 = blockIdx.x * BTILE;     // global tile base row

  // ---- Phase 0: stage W_hh -> LDS f16, zero h ------------------------------
  for (int i = tid; i < G3 * HID; i += NTHREADS) sWhh[i] = (_Float16)W_hh[i];
  for (int i = tid; i < BTILE * HID; i += NTHREADS) sH[i] = (_Float16)0.f;
  __syncthreads();

  // ---- Phase 1: sGpT = f16( x @ W_ih^T + b_ih (+ b_hh for r,z cols) )^T ----
  {
    v16h xa[8];                             // A-frags of this strip's x rows (K=256)
#pragma unroll
    for (int s = 0; s < 8; ++s)
      xa[s] = frag_gf32(x + (size_t)(bRow0 + lr0 + nl) * DIN, 32 * s, koff);

#pragma unroll 1
    for (int jn = 0; jn < G3 / 16; ++jn) {  // 24 output column tiles
      v8f acc = {};
#pragma unroll
      for (int k = 0; k < 8; ++k) {
        v16h bf = frag_gf32(W_ih + (size_t)(jn * 16 + nl) * DIN, 32 * k, koff);
        acc = wmma_f32_f16(xa[k], bf, acc);
      }
      const int col  = jn * 16 + nl;
      const float bias = b_ih[col] + ((col < 2 * HID) ? b_hh[col] : 0.f);
      v8h gph;                              // rows are contiguous in sGpT -> one b128
#pragma unroll
      for (int vr = 0; vr < 8; ++vr) gph[vr] = (_Float16)(acc[vr] + bias);
      *(v8h*)(sGpT + (size_t)col * BTILE + lr0 + koff) = gph;
    }
  }
  __syncthreads();

  // ---- per-lane constants for the output head / n-gate bias ----------------
  float wo0[8], wo1[8], wo2[8], bhn[8];
#pragma unroll
  for (int j = 0; j < 8; ++j) {
    const int col = j * 16 + nl;
    wo0[j] = W_out[0 * HID + col];
    wo1[j] = W_out[1 * HID + col];
    wo2[j] = W_out[2 * HID + col];
    bhn[j] = b_hh[2 * HID + col];
  }
  const float bo0 = b_out[0], bo1 = b_out[1], bo2 = b_out[2];

  // h kept in registers in C/D layout: hreg[j][vr] = h[lr0+vr+koff, j*16+nl]
  v8h hreg[8] = {};

  // ---- Phase 2: 60-step recurrence, fully on-chip --------------------------
#pragma unroll 1
  for (int t = 0; t < HORIZON; ++t) {
    v16h ha[4];                             // A-frags of h (K=128)
#pragma unroll
    for (int s = 0; s < 4; ++s)
      ha[s] = frag_lds(sH + (lr0 + nl) * HID, 32 * s, koff);

    float po0[8] = {0.f}, po1[8] = {0.f}, po2[8] = {0.f};

    // FULLY UNROLLED: all per-j arrays (hreg/wo*/bhn) get constant indices and
    // stay in VGPRs -> no scratch spill traffic in the hot loop.
#pragma unroll
    for (int j = 0; j < 8; ++j) {           // 8 column tiles of the 128 h-cols
      v8f aR = {}, aZ = {}, aN = {};
#pragma unroll
      for (int k = 0; k < 4; ++k) {
        v16h bf;
        bf = frag_lds(sWhh + (size_t)(0 * HID + j * 16 + nl) * HID, 32 * k, koff);
        aR = wmma_f32_f16(ha[k], bf, aR);
        bf = frag_lds(sWhh + (size_t)(1 * HID + j * 16 + nl) * HID, 32 * k, koff);
        aZ = wmma_f32_f16(ha[k], bf, aZ);
        bf = frag_lds(sWhh + (size_t)(2 * HID + j * 16 + nl) * HID, 32 * k, koff);
        aN = wmma_f32_f16(ha[k], bf, aN);
      }
      const int col = j * 16 + nl;
      // vector gp loads: 8 consecutive rows per lane from transposed layout
      const v8h gr = *(const v8h*)(sGpT + (size_t)(0 * HID + col) * BTILE + lr0 + koff);
      const v8h gz = *(const v8h*)(sGpT + (size_t)(1 * HID + col) * BTILE + lr0 + koff);
      const v8h gn = *(const v8h*)(sGpT + (size_t)(2 * HID + col) * BTILE + lr0 + koff);
#pragma unroll
      for (int vr = 0; vr < 8; ++vr) {
        const float rg = sigmoid_((float)gr[vr] + aR[vr]);
        const float zg = sigmoid_((float)gz[vr] + aZ[vr]);
        const float ng = tanh_((float)gn[vr] + rg * (aN[vr] + bhn[j]));
        const float hn = (1.f - zg) * ng + zg * (float)hreg[j][vr];
        hreg[j][vr] = (_Float16)hn;
        // transpose path for next step's A-fragments (row-major h tile)
        sH[(lr0 + vr + koff) * HID + col] = (_Float16)hn;
        po0[vr] = fmaf(hn, wo0[j], po0[vr]);
        po1[vr] = fmaf(hn, wo1[j], po1[vr]);
        po2[vr] = fmaf(hn, wo2[j], po2[vr]);
      }
    }

    // reduce 3-wide head over the 16 columns held across lanes (groups of 16)
#pragma unroll
    for (int vr = 0; vr < 8; ++vr) {
      float s0 = po0[vr], s1 = po1[vr], s2 = po2[vr];
#pragma unroll
      for (int m = 1; m < 16; m <<= 1) {
        s0 += __shfl_xor(s0, m, 16);
        s1 += __shfl_xor(s1, m, 16);
        s2 += __shfl_xor(s2, m, 16);
      }
      if (nl == 0) {
        const int rowg = bRow0 + lr0 + vr + koff;
        float* op = out + ((size_t)rowg * HORIZON + t) * 3;
        op[0] = s0 + bo0; op[1] = s1 + bo1; op[2] = s2 + bo2;
      }
    }
    // strips are wave-private; LDS is in-order per wave -> no block barrier.
    // Compiler-only fence to keep the sH stores ordered vs next A-frag loads.
    __builtin_amdgcn_wave_barrier();
  }
}

extern "C" void kernel_launch(void* const* d_in, const int* in_sizes, int n_in,
                              void* d_out, int out_size, void* d_ws, size_t ws_size,
                              hipStream_t stream) {
  const float* x     = (const float*)d_in[0];
  const float* W_ih  = (const float*)d_in[1];
  const float* W_hh  = (const float*)d_in[2];
  const float* b_ih  = (const float*)d_in[3];
  const float* b_hh  = (const float*)d_in[4];
  const float* W_out = (const float*)d_in[5];
  const float* b_out = (const float*)d_in[6];
  float* out = (float*)d_out;

  const int B      = in_sizes[0] / DIN;         // 16384
  const int blocks = B / BTILE;                 // 256
  const size_t lds = (size_t)(G3 * HID + G3 * BTILE + BTILE * HID) * sizeof(_Float16); // 160 KB

  gru_traj_kernel<<<blocks, NTHREADS, lds, stream>>>(x, W_ih, W_hh, b_ih, b_hh,
                                                     W_out, b_out, out);
}
